// HyperCHARMModel_28183575396907
// MI455X (gfx1250) — matrix operand order repaired and checked by reading, not compiled, not run on hardware
//
#include <hip/hip_runtime.h>
#include <hip/hip_bf16.h>

#define N_NODES   100000
#define M_EDGES   50000
#define E_INC     1000000
#define NODE_DIM  768
#define HID       128

typedef __attribute__((ext_vector_type(16))) __bf16 v16bf;
typedef __attribute__((ext_vector_type(8)))  __bf16 v8bf;
typedef __attribute__((ext_vector_type(4)))  __bf16 v4bf;
typedef __attribute__((ext_vector_type(8)))  float  v8f;

// ---------------------------------------------------------------------------
// WMMA helper: D = A(16x32 bf16) * B(32x16 bf16) + C(16x16 f32)
// ---------------------------------------------------------------------------
__device__ __forceinline__ v8f wmma_bf16(v16bf a, v16bf b, v8f c) {
    return __builtin_amdgcn_wmma_f32_16x16x32_bf16(false, a, false, b,
                                                   (short)0, c, false, false);
}

// A fragment from a bf16 row in LDS.  Per documented 16-bit A layout:
// lane(lo half): K = base+{0..7, 16..23}; lane(hi half): K = base+{8..15, 24..31}
// caller passes p = row + c*32 + hi*8  ->  elems 0..7 = p[0..7], 8..15 = p[16..23]
__device__ __forceinline__ v16bf load_afrag(const __bf16* p) {
    v8bf lo = *(const v8bf*)p;
    v8bf hi = *(const v8bf*)(p + 16);
    v16bf r;
#pragma unroll
    for (int i = 0; i < 8; ++i) { r[i] = lo[i]; r[i + 8] = hi[i]; }
    return r;
}

// A fragment directly from f32 global memory (input projection), b128 loads
__device__ __forceinline__ v16bf load_afrag_f32(const float* p) {
    float4 a = *(const float4*)(p + 0);
    float4 b = *(const float4*)(p + 4);
    float4 c = *(const float4*)(p + 16);
    float4 d = *(const float4*)(p + 20);
    v16bf r;
    r[0] = (__bf16)a.x;  r[1] = (__bf16)a.y;  r[2]  = (__bf16)a.z;  r[3]  = (__bf16)a.w;
    r[4] = (__bf16)b.x;  r[5] = (__bf16)b.y;  r[6]  = (__bf16)b.z;  r[7]  = (__bf16)b.w;
    r[8] = (__bf16)c.x;  r[9] = (__bf16)c.y;  r[10] = (__bf16)c.z;  r[11] = (__bf16)c.w;
    r[12] = (__bf16)d.x; r[13] = (__bf16)d.y; r[14] = (__bf16)d.z;  r[15] = (__bf16)d.w;
    return r;
}

// B fragment: pre-packed contiguous 16 bf16 per lane (two b128 loads)
__device__ __forceinline__ v16bf load_bfrag(const __bf16* p) {
    v8bf lo = *(const v8bf*)p;
    v8bf hi = *(const v8bf*)(p + 8);
    v16bf r;
#pragma unroll
    for (int i = 0; i < 8; ++i) { r[i] = lo[i]; r[i + 8] = hi[i]; }
    return r;
}

__device__ __forceinline__ v4bf pack4(float a, float b, float c, float d) {
    v4bf o;
    o[0] = (__bf16)a; o[1] = (__bf16)b; o[2] = (__bf16)c; o[3] = (__bf16)d;
    return o;
}

// ---------------------------------------------------------------------------
// Weight packer: W [K x 128] f32 row-major -> bf16 fragments.
// Fragment element j of lane L in (chunk c, coltile t):
//   k = c*32 + (L>>4)*8 + (j&7) + (j>>3)*16 ,  n = t*16 + (L&15)   (zero-padded past K)
// ---------------------------------------------------------------------------
__global__ void pack_weight_kernel(const float* __restrict__ W, int K, int chunks,
                                   __bf16* __restrict__ dst) {
    int total = chunks * 4096;                 // chunks * 8 tiles * 32 lanes * 16
    int idx = blockIdx.x * 256 + threadIdx.x;
    if (idx >= total) return;
    int j    = idx & 15;
    int lane = (idx >> 4) & 31;
    int t    = (idx >> 9) & 7;
    int c    = idx >> 12;
    int k = c * 32 + (lane >> 4) * 8 + (j & 7) + (j >> 3) * 16;
    int n = t * 16 + (lane & 15);
    float v = (k < K) ? W[(long)k * 128 + n] : 0.0f;
    dst[idx] = (__bf16)v;
}

// ---------------------------------------------------------------------------
// degree = bincount(node_ids)
// ---------------------------------------------------------------------------
__global__ void deg_kernel(const int* __restrict__ nodeIds, float* __restrict__ deg) {
    int tid = blockIdx.x * 256 + threadIdx.x;
    if (tid < E_INC) atomicAdd(&deg[nodeIds[tid]], 1.0f);
}

// ---------------------------------------------------------------------------
// h = relu(x @ in_w + in_b).  1 wave = 16 rows, K=768 (24 chunks), Nout=128.
// ---------------------------------------------------------------------------
__global__ __launch_bounds__(128) void input_proj_kernel(
    const float* __restrict__ x, const __bf16* __restrict__ packW,
    const float* __restrict__ in_b, float* __restrict__ hbuf) {
    int w = threadIdx.x >> 5, lane = threadIdx.x & 31;
    int low = lane & 15, hi = lane >> 4;
    int tile = blockIdx.x * 4 + w;
    if (tile >= N_NODES / 16) return;
    long base = (long)tile * 16;

    v8f zero = {0, 0, 0, 0, 0, 0, 0, 0};
    v8f acc[8];
#pragma unroll
    for (int t = 0; t < 8; ++t) acc[t] = zero;

    const float* xrow = x + (base + low) * NODE_DIM;
    for (int c = 0; c < NODE_DIM / 32; ++c) {
        v16bf af = load_afrag_f32(xrow + c * 32 + hi * 8);
        v16bf bfr[8];
#pragma unroll
        for (int t = 0; t < 8; ++t)
            bfr[t] = load_bfrag(packW + ((long)(c * 8 + t) * 512 + lane * 16));
#pragma unroll
        for (int t = 0; t < 8; ++t) acc[t] = wmma_bf16(af, bfr[t], acc[t]);
    }
    // C layout: elem v of acc[t] -> row = v + hi*8, col = t*16 + low
#pragma unroll
    for (int t = 0; t < 8; ++t) {
        int col = t * 16 + low;
        float b = in_b[col];
#pragma unroll
        for (int v = 0; v < 8; ++v) {
            long r = base + v + hi * 8;
            hbuf[r * HID + col] = fmaxf(acc[t][v] + b, 0.0f);
        }
    }
}

// ---------------------------------------------------------------------------
// Fused edge MLP: gather -> GEMM1 -> bias+LN+ReLU (in regs) -> GEMM2 -> bias
// (-> ReLU for MODE 1) -> f32 atomic segment-sum scatter.
// MODE 0 (n2e): A = [h[node] | he_mark[he]] (K=130, KPAD=160), scatter by he.
// MODE 1 (e2n): A = [he_attr[he] | agg[he]/cnt] (K=192),       scatter by node.
// 1 wave = 16 incidences, 4 waves/block.
// ---------------------------------------------------------------------------
template <int MODE, int KPAD, int KCH1>
__global__ __launch_bounds__(128) void edge_mlp_kernel(
    const int* __restrict__ nodeIds, const int* __restrict__ heIds,
    const float* __restrict__ hbuf,     // MODE 0
    const float* __restrict__ heMark,   // MODE 0
    const float* __restrict__ heAttr,   // MODE 1
    const float* __restrict__ aggBuf,   // MODE 1
    const float* __restrict__ heCount,  // MODE 1
    const __bf16* __restrict__ packW1, const __bf16* __restrict__ packW2,
    const float* __restrict__ b1, const float* __restrict__ g1,
    const float* __restrict__ be1, const float* __restrict__ b2,
    float* __restrict__ dst) {
    __shared__ __align__(16) __bf16 aT[4][16][KPAD];
    __shared__ __align__(16) __bf16 a2[4][16][HID];
    __shared__ int sid[4][16];

    int w = threadIdx.x >> 5, lane = threadIdx.x & 31;
    int low = lane & 15, hi = lane >> 4;
    int tile = blockIdx.x * 4 + w;          // always < E/16 (exact division)
    int e = tile * 16 + low;

    // ---- gather A tile (both lane-halves share row `low`), float4-vectorized ----
    if (MODE == 0) {
        int nid = nodeIds[e], hid = heIds[e];
        if (hi == 0) sid[w][low] = hid;
        const float* hp = hbuf + (long)nid * HID;
#pragma unroll
        for (int j = 0; j < 16; ++j) {
            int c = hi * 64 + j * 4;
            float4 f = *(const float4*)(hp + c);
            *(v4bf*)&aT[w][low][c] = pack4(f.x, f.y, f.z, f.w);
        }
        if (hi) {
            float m0 = heMark[(long)hid * 2 + 0];
            float m1 = heMark[(long)hid * 2 + 1];
            *(v4bf*)&aT[w][low][128] = pack4(m0, m1, 0.0f, 0.0f);
#pragma unroll
            for (int c = 132; c < KPAD; c += 4)
                *(v4bf*)&aT[w][low][c] = pack4(0.0f, 0.0f, 0.0f, 0.0f);
        }
    } else {
        int hid = heIds[e], nid = nodeIds[e];
        if (hi == 0) sid[w][low] = nid;
        float inv = 1.0f / (heCount[hid] + 1e-6f);
        const float* ap = aggBuf + (long)hid * HID;
        const float* ep = heAttr + (long)hid * 64;
#pragma unroll
        for (int j = 0; j < 24; ++j) {
            int c = hi * 96 + j * 4;   // [0,64): he_attr ; [64,192): agg * inv
            float4 f;
            if (c < 64) {
                f = *(const float4*)(ep + c);
            } else {
                f = *(const float4*)(ap + (c - 64));
                f.x *= inv; f.y *= inv; f.z *= inv; f.w *= inv;
            }
            *(v4bf*)&aT[w][low][c] = pack4(f.x, f.y, f.z, f.w);
        }
    }
    __syncthreads();

    // per-lane column params (col = t*16 + low)
    float b1v[8], g1v[8], be1v[8], b2v[8];
#pragma unroll
    for (int t = 0; t < 8; ++t) {
        int col = t * 16 + low;
        b1v[t] = b1[col]; g1v[t] = g1[col]; be1v[t] = be1[col]; b2v[t] = b2[col];
    }

    v8f zero = {0, 0, 0, 0, 0, 0, 0, 0};
    v8f acc[8];
#pragma unroll
    for (int t = 0; t < 8; ++t) acc[t] = zero;

    // ---- GEMM1: [16 x KPAD] @ [KPAD x 128] ----
    // group all 8 B-fragment loads before the 8 WMMAs so the compiler can
    // clause them and overlap chunk c+1 loads with chunk c WMMAs
#pragma unroll
    for (int c = 0; c < KCH1; ++c) {
        v16bf af = load_afrag(&aT[w][low][c * 32 + hi * 8]);
        v16bf bfr[8];
#pragma unroll
        for (int t = 0; t < 8; ++t)
            bfr[t] = load_bfrag(packW1 + ((long)(c * 8 + t) * 512 + lane * 16));
#pragma unroll
        for (int t = 0; t < 8; ++t) acc[t] = wmma_bf16(af, bfr[t], acc[t]);
    }

    // ---- bias + LayerNorm(128) + ReLU, all in registers ----
    float s1[8], s2[8];
#pragma unroll
    for (int v = 0; v < 8; ++v) { s1[v] = 0.0f; s2[v] = 0.0f; }
#pragma unroll
    for (int t = 0; t < 8; ++t)
#pragma unroll
        for (int v = 0; v < 8; ++v) {
            float f = acc[t][v] + b1v[t];
            acc[t][v] = f;
            s1[v] += f; s2[v] += f * f;
        }
    // each row lives in 16 lanes of one half; xor-masks <16 stay in the half
#pragma unroll
    for (int m = 1; m < 16; m <<= 1)
#pragma unroll
        for (int v = 0; v < 8; ++v) {
            s1[v] += __shfl_xor(s1[v], m, 32);
            s2[v] += __shfl_xor(s2[v], m, 32);
        }
    float mean[8], rstd[8];
#pragma unroll
    for (int v = 0; v < 8; ++v) {
        mean[v] = s1[v] * (1.0f / 128.0f);
        float var = s2[v] * (1.0f / 128.0f) - mean[v] * mean[v];
        rstd[v] = rsqrtf(var + 1e-5f);
    }
#pragma unroll
    for (int t = 0; t < 8; ++t)
#pragma unroll
        for (int v = 0; v < 8; ++v) {
            float f = (acc[t][v] - mean[v]) * rstd[v] * g1v[t] + be1v[t];
            f = fmaxf(f, 0.0f);
            a2[w][v + hi * 8][t * 16 + low] = (__bf16)f;
        }
    __syncthreads();

    // ---- GEMM2: [16 x 128] @ [128 x 128] ----
    v8f acc2[8];
#pragma unroll
    for (int t = 0; t < 8; ++t) acc2[t] = zero;
#pragma unroll
    for (int c = 0; c < 4; ++c) {
        v16bf af = load_afrag(&a2[w][low][c * 32 + hi * 8]);
        v16bf bfr[8];
#pragma unroll
        for (int t = 0; t < 8; ++t)
            bfr[t] = load_bfrag(packW2 + ((long)(c * 8 + t) * 512 + lane * 16));
#pragma unroll
        for (int t = 0; t < 8; ++t) acc2[t] = wmma_bf16(af, bfr[t], acc2[t]);
    }

    // ---- bias (+ReLU for e2n) and atomic segment-sum scatter ----
#pragma unroll
    for (int t = 0; t < 8; ++t) {
        int col = t * 16 + low;
#pragma unroll
        for (int v = 0; v < 8; ++v) {
            int row = v + hi * 8;
            float val = acc2[t][v] + b2v[t];
            if (MODE == 1) val = fmaxf(val, 0.0f);
            atomicAdd(&dst[(long)sid[w][row] * HID + col], val);
        }
    }
}

// ---------------------------------------------------------------------------
// h += LayerNorm(out / (deg+1e-6)) * g + b     (1 wave per node row)
// ---------------------------------------------------------------------------
__global__ __launch_bounds__(256) void node_update_kernel(
    const float* __restrict__ outBuf, const float* __restrict__ deg,
    const float* __restrict__ lng, const float* __restrict__ lnb,
    float* __restrict__ hbuf) {
    int w = threadIdx.x >> 5, lane = threadIdx.x & 31;
    long r = (long)blockIdx.x * 8 + w;
    float dinv = 1.0f / (deg[r] + 1e-6f);
    const float* op = outBuf + r * HID;
    float4 f = *(const float4*)(op + lane * 4);
    float v[4] = {f.x * dinv, f.y * dinv, f.z * dinv, f.w * dinv};
    float s1 = v[0] + v[1] + v[2] + v[3];
    float s2 = v[0]*v[0] + v[1]*v[1] + v[2]*v[2] + v[3]*v[3];
#pragma unroll
    for (int m = 1; m < 32; m <<= 1) {
        s1 += __shfl_xor(s1, m, 32);
        s2 += __shfl_xor(s2, m, 32);
    }
    float mean = s1 * (1.0f / 128.0f);
    float var  = s2 * (1.0f / 128.0f) - mean * mean;
    float rstd = rsqrtf(var + 1e-5f);
    float4 g = *(const float4*)(lng + lane * 4);
    float4 b = *(const float4*)(lnb + lane * 4);
    float4* hp4 = (float4*)(hbuf + r * HID + lane * 4);
    float4 h4 = *hp4;
    h4.x += (v[0] - mean) * rstd * g.x + b.x;
    h4.y += (v[1] - mean) * rstd * g.y + b.y;
    h4.z += (v[2] - mean) * rstd * g.z + b.z;
    h4.w += (v[3] - mean) * rstd * g.w + b.w;
    *hp4 = h4;
}

// ---------------------------------------------------------------------------
// pred head: Linear(128,64)+bias -> LN(64) -> ReLU -> Linear(64,1)+bias
// 1 wave per node row; each lane owns 2 of the 64 hidden features.
// ---------------------------------------------------------------------------
__global__ __launch_bounds__(256) void pred_kernel(
    const float* __restrict__ h, const float* __restrict__ w1,
    const float* __restrict__ b1, const float* __restrict__ g1,
    const float* __restrict__ be1, const float* __restrict__ w2,
    const float* __restrict__ b2, float* __restrict__ outv) {
    int w = threadIdx.x >> 5, lane = threadIdx.x & 31;
    long r = (long)blockIdx.x * 8 + w;
    const float* hp = h + r * HID;
    float a0 = 0.0f, a1 = 0.0f;
    for (int c = 0; c < HID; ++c) {
        float hv = hp[c];
        a0 += hv * w1[c * 64 + lane];
        a1 += hv * w1[c * 64 + lane + 32];
    }
    a0 += b1[lane]; a1 += b1[lane + 32];
    float s1 = a0 + a1, s2 = a0 * a0 + a1 * a1;
#pragma unroll
    for (int m = 1; m < 32; m <<= 1) {
        s1 += __shfl_xor(s1, m, 32);
        s2 += __shfl_xor(s2, m, 32);
    }
    float mean = s1 * (1.0f / 64.0f);
    float var  = s2 * (1.0f / 64.0f) - mean * mean;
    float rstd = rsqrtf(var + 1e-5f);
    float n0 = fmaxf((a0 - mean) * rstd * g1[lane] + be1[lane], 0.0f);
    float n1 = fmaxf((a1 - mean) * rstd * g1[lane + 32] + be1[lane + 32], 0.0f);
    float d = n0 * w2[lane] + n1 * w2[lane + 32];
#pragma unroll
    for (int m = 1; m < 32; m <<= 1) d += __shfl_xor(d, m, 32);
    if (lane == 0) outv[r] = d + b2[0];
}

// ---------------------------------------------------------------------------
// host launcher
// ---------------------------------------------------------------------------
extern "C" void kernel_launch(void* const* d_in, const int* in_sizes, int n_in,
                              void* d_out, int out_size, void* d_ws, size_t ws_size,
                              hipStream_t stream) {
    (void)in_sizes; (void)n_in; (void)out_size; (void)ws_size;

    const float* x       = (const float*)d_in[0];
    const int*   heidx   = (const int*)d_in[1];
    const int*   nodeIds = heidx;
    const int*   heIdsP  = heidx + E_INC;
    const float* heAttr  = (const float*)d_in[2];
    const float* heMark  = (const float*)d_in[3];
    const float* heCount = (const float*)d_in[4];
    // params flattened in sorted-key pytree order:
    const float* in_b = (const float*)d_in[5];
    const float* in_w = (const float*)d_in[6];
    // per layer (base=7+14*l): e2n{b1,b2,be1,g1,w1,w2}, ln_b, ln_g, n2e{b1,b2,be1,g1,w1,w2}
    const float* pred_b1  = (const float*)d_in[35];
    const float* pred_b2  = (const float*)d_in[36];
    const float* pred_be1 = (const float*)d_in[37];
    const float* pred_g1  = (const float*)d_in[38];
    const float* pred_w1  = (const float*)d_in[39];
    const float* pred_w2  = (const float*)d_in[40];

    // --- workspace carve-out (256B aligned) ---
    char* ws = (char*)d_ws;
    size_t cur = 0;
    auto alloc = [&](size_t bytes) -> char* {
        char* p = ws + cur;
        cur = (cur + bytes + 255) & ~(size_t)255;
        return p;
    };
    float*  hbuf   = (float*)alloc((size_t)N_NODES * HID * 4);
    float*  degbuf = (float*)alloc((size_t)N_NODES * 4);
    float*  aggbuf = (float*)alloc((size_t)M_EDGES * HID * 4);
    float*  outbuf = (float*)alloc((size_t)N_NODES * HID * 4);
    __bf16* packIn = (__bf16*)alloc((size_t)24 * 4096 * 2);   // in_w: 24 chunks
    __bf16* pN2E1[2], *pN2E2[2], *pE2N1[2], *pE2N2[2];
    for (int l = 0; l < 2; ++l) {
        pN2E1[l] = (__bf16*)alloc((size_t)5 * 4096 * 2);      // K=130 -> 5 chunks
        pN2E2[l] = (__bf16*)alloc((size_t)4 * 4096 * 2);      // K=128 -> 4 chunks
        pE2N1[l] = (__bf16*)alloc((size_t)6 * 4096 * 2);      // K=192 -> 6 chunks
        pE2N2[l] = (__bf16*)alloc((size_t)4 * 4096 * 2);      // K=128 -> 4 chunks
    }

    // --- pack all GEMM weights to bf16 fragments ---
    auto pack = [&](const float* W, int K, int chunks, __bf16* dst) {
        int total = chunks * 4096;
        pack_weight_kernel<<<(total + 255) / 256, 256, 0, stream>>>(W, K, chunks, dst);
    };
    pack(in_w, NODE_DIM, 24, packIn);
    for (int l = 0; l < 2; ++l) {
        int base = 7 + 14 * l;
        pack((const float*)d_in[base + 12], HID + 2, 5, pN2E1[l]); // n2e.w1 [130,128]
        pack((const float*)d_in[base + 13], HID,     4, pN2E2[l]); // n2e.w2 [128,128]
        pack((const float*)d_in[base + 4],  64 + HID,6, pE2N1[l]); // e2n.w1 [192,128]
        pack((const float*)d_in[base + 5],  HID,     4, pE2N2[l]); // e2n.w2 [128,128]
    }

    // --- degree bincount ---
    hipMemsetAsync(degbuf, 0, (size_t)N_NODES * 4, stream);
    deg_kernel<<<(E_INC + 255) / 256, 256, 0, stream>>>(nodeIds, degbuf);

    // --- input projection ---
    input_proj_kernel<<<(N_NODES / 16 + 3) / 4, 128, 0, stream>>>(x, packIn, in_b, hbuf);

    // --- layers ---
    const int edgeBlocks = (E_INC / 16) / 4;   // 62500 tiles / 4 waves = 15625
    for (int l = 0; l < 2; ++l) {
        int base = 7 + 14 * l;
        const float* e2n_b1  = (const float*)d_in[base + 0];
        const float* e2n_b2  = (const float*)d_in[base + 1];
        const float* e2n_be1 = (const float*)d_in[base + 2];
        const float* e2n_g1  = (const float*)d_in[base + 3];
        const float* ln_b    = (const float*)d_in[base + 6];
        const float* ln_g    = (const float*)d_in[base + 7];
        const float* n2e_b1  = (const float*)d_in[base + 8];
        const float* n2e_b2  = (const float*)d_in[base + 9];
        const float* n2e_be1 = (const float*)d_in[base + 10];
        const float* n2e_g1  = (const float*)d_in[base + 11];

        hipMemsetAsync(aggbuf, 0, (size_t)M_EDGES * HID * 4, stream);
        edge_mlp_kernel<0, 160, 5><<<edgeBlocks, 128, 0, stream>>>(
            nodeIds, heIdsP, hbuf, heMark, nullptr, nullptr, nullptr,
            pN2E1[l], pN2E2[l], n2e_b1, n2e_g1, n2e_be1, n2e_b2, aggbuf);

        hipMemsetAsync(outbuf, 0, (size_t)N_NODES * HID * 4, stream);
        edge_mlp_kernel<1, 192, 6><<<edgeBlocks, 128, 0, stream>>>(
            nodeIds, heIdsP, nullptr, nullptr, heAttr, aggbuf, heCount,
            pE2N1[l], pE2N2[l], e2n_b1, e2n_g1, e2n_be1, e2n_b2, outbuf);

        node_update_kernel<<<N_NODES / 8, 256, 0, stream>>>(outbuf, degbuf, ln_g, ln_b, hbuf);
    }

    // --- prediction head ---
    pred_kernel<<<N_NODES / 8, 256, 0, stream>>>(hbuf, pred_w1, pred_b1, pred_g1,
                                                 pred_be1, pred_w2, pred_b2,
                                                 (float*)d_out);
}
